// ModifiedVAETransformer_82446192214372
// MI455X (gfx1250) — matrix operand order repaired and checked by reading, not compiled, not run on hardware
//
// CDNA5 / gfx1250 implementation of the VAE-transformer forward pass.
// All heavy GEMMs use V_WMMA_F32_16X16X32_BF16 (bf16 in, f32 accum); fragments
// are loaded straight from global memory in the ISA-native register layouts
// (no LDS transpose needed for row-major A and row-major B). The K-loop is
// 2x-unrolled with two named fragment sets (copy-free double buffering): each
// WMMA group overlaps the other set's loads (4 b128 for A, 64 16-bit for B), with no register rotation
// movs. K % 64 == 0 holds for every GEMM in this model (128 / 768 / 3072).
#include <hip/hip_runtime.h>
#include <hip/hip_bf16.h>

#define EMB   768
#define NHEAD 6
#define HDIM  128
#define VOCAB 8192
#define LSEQ  10
#define LATD  12
#define NB    2048
#define FFD   (4 * EMB)            // 3072
#define NLROWS (NB * LSEQ)         // 20480
#define NLE   ((size_t)NLROWS * EMB)
#define LN_EPS 1e-5f

typedef __bf16 bf16;
typedef __attribute__((ext_vector_type(16))) __bf16 bf16x16;
typedef __attribute__((ext_vector_type(8)))  float  f32x8;
typedef __attribute__((ext_vector_type(4)))  unsigned int u32x4;

__device__ __forceinline__ bf16x16 make_frag(u32x4 lo, u32x4 hi) {
  struct P { u32x4 a, b; } p{lo, hi};
  return __builtin_bit_cast(bf16x16, p);
}

// Load one K-tile worth of fragments (2 A frags for the 32-row wave tile,
// 4 B frags for the 64-col wave tile) in the ISA-native register striping.
__device__ __forceinline__ void load_frags(const bf16* __restrict__ a0,
                                           const bf16* __restrict__ a1,
                                           const bf16* __restrict__ brow,
                                           int k0, int N,
                                           bf16x16 fa[2], bf16x16 fb[4]) {
  u32x4 lo0 = *(const u32x4*)(a0 + k0);
  u32x4 hi0 = *(const u32x4*)(a0 + k0 + 16);
  u32x4 lo1 = *(const u32x4*)(a1 + k0);
  u32x4 hi1 = *(const u32x4*)(a1 + k0 + 16);
  fa[0] = make_frag(lo0, hi0);
  fa[1] = make_frag(lo1, hi1);
#pragma unroll
  for (int j = 0; j < 4; ++j) {
    bf16x16 v;
#pragma unroll
    for (int s = 0; s < 16; ++s)
      v[s] = brow[(size_t)(k0 + (s < 8 ? s : s + 8)) * N + j * 16];
    fb[j] = v;
  }
}

__device__ __forceinline__ void wmma8(const bf16x16 fa[2], const bf16x16 fb[4],
                                      f32x8 acc[2][4]) {
#pragma unroll
  for (int i = 0; i < 2; ++i)
#pragma unroll
    for (int j = 0; j < 4; ++j)
      acc[i][j] = __builtin_amdgcn_wmma_f32_16x16x32_bf16(
          false, fa[i], false, fb[j], (short)0, acc[i][j], false, false);
}

// ---------------------------------------------------------------------------
// Tiled bf16 GEMM: C[MxN] = A[MxK] * B[KxN] (+bias[N]) , optional ReLU.
// Block = WM*WN waves (wave32). Each wave computes a 32x64 tile = 2x4 WMMAs.
// A row-major bf16, B row-major bf16, C row-major OutT (float or bf16).
// M % (WM*32) == 0, N % (WN*64) == 0, K % 64 == 0 (guaranteed by launches).
// ---------------------------------------------------------------------------
template <int WM, int WN, bool RELU, typename OutT>
__global__ __launch_bounds__(WM * WN * 32)
void gemm_bf16_kernel(const bf16* __restrict__ A, const bf16* __restrict__ B,
                      const float* __restrict__ bias, OutT* __restrict__ C,
                      int M, int K, int N) {
  const int lane = threadIdx.x & 31;
  const int wave = threadIdx.x >> 5;
  const int wm = wave / WN, wn = wave % WN;
  const int mB = (blockIdx.y * WM + wm) * 32;
  const int nB = (blockIdx.x * WN + wn) * 64;
  const int ncol = lane & 15;

  f32x8 acc[2][4];
#pragma unroll
  for (int j = 0; j < 4; ++j) {
    float bv = bias ? bias[nB + j * 16 + ncol] : 0.0f;
    f32x8 c;
#pragma unroll
    for (int r = 0; r < 8; ++r) c[r] = bv;
    acc[0][j] = c;
    acc[1][j] = c;
  }

  // A fragment addressing (16-bit A 16x32 layout, ISA 7.12.2):
  //   lanes 0-15: row m=lane,   Ks {0..7, 16..23}
  //   lanes16-31: row m=lane-16,Ks {8..15,24..31}
  const int mrow = lane & 15;
  const int aoff = (lane < 16) ? 0 : 8;
  const bf16* a0 = A + (size_t)(mB + mrow) * K + aoff;
  const bf16* a1 = a0 + (size_t)16 * K;
  // B fragment addressing (16-bit B 32x16): lane holds col n=lane%16, Ks same as A.
  const bf16* brow = B + (size_t)((lane >> 4) * 8) * N + nB + (lane & 15);

  // Copy-free double buffering: two named fragment sets alternate roles in a
  // 2x-unrolled K loop; each WMMA group overlaps the other set's loads.
  bf16x16 fa0[2], fb0[4], fa1[2], fb1[4];
  load_frags(a0, a1, brow, 0, N, fa0, fb0);
  load_frags(a0, a1, brow, 32, N, fa1, fb1);
  for (int k0 = 64; k0 < K; k0 += 64) {
    // Warm L2 one step ahead (speculative; stays inside the ws arena).
    __builtin_prefetch(a0 + k0 + 64, 0, 1);
    __builtin_prefetch(brow + (size_t)(k0 + 64) * N, 0, 1);
    wmma8(fa0, fb0, acc);                         // tile k0-64
    load_frags(a0, a1, brow, k0, N, fa0, fb0);    // tile k0
    wmma8(fa1, fb1, acc);                         // tile k0-32
    load_frags(a0, a1, brow, k0 + 32, N, fa1, fb1);  // tile k0+32
  }
  wmma8(fa0, fb0, acc);                           // tile K-64
  wmma8(fa1, fb1, acc);                           // tile K-32

  // C/D layout: lane l -> col n = l%16; VGPR r -> row r + (l<16 ? 0 : 8).
  const int mr0 = mB + ((lane < 16) ? 0 : 8);
#pragma unroll
  for (int i = 0; i < 2; ++i)
#pragma unroll
    for (int j = 0; j < 4; ++j)
#pragma unroll
      for (int r = 0; r < 8; ++r) {
        float v = acc[i][j][r];
        if (RELU) v = fmaxf(v, 0.0f);
        C[(size_t)(mr0 + i * 16 + r) * N + nB + j * 16 + ncol] = (OutT)v;
      }
}

// ---------------------------------------------------------------------------
// Small / elementwise kernels (fp32 VALU; negligible FLOPs)
// ---------------------------------------------------------------------------
__global__ __launch_bounds__(256)
void f2b_kernel(const float* __restrict__ in, bf16* __restrict__ out, size_t n) {
  size_t i = (size_t)blockIdx.x * 256 + threadIdx.x;
  if (i < n) out[i] = (bf16)in[i];
}

__global__ __launch_bounds__(256)
void embed_enc_kernel(const int* __restrict__ x, const float* __restrict__ we,
                      const float* __restrict__ pe, float* __restrict__ hf,
                      bf16* __restrict__ hb) {
  size_t i = (size_t)blockIdx.x * 256 + threadIdx.x;
  if (i >= NLE) return;
  size_t row = i / EMB;
  int e = (int)(i % EMB);
  int l = (int)(row % LSEQ);
  float v = we[(size_t)x[row] * EMB + e] + pe[(size_t)l * EMB + e];
  hf[i] = v;
  hb[i] = (bf16)v;
}

__global__ __launch_bounds__(256)
void dec_embed_kernel(const int* __restrict__ tok, const float* __restrict__ we,
                      const float* __restrict__ pe, int p, bf16* __restrict__ xe) {
  size_t i = (size_t)blockIdx.x * 256 + threadIdx.x;
  if (i >= (size_t)NB * EMB) return;
  int n = (int)(i / EMB), e = (int)(i % EMB);
  xe[i] = (bf16)(we[(size_t)tok[n] * EMB + e] + pe[(size_t)p * EMB + e]);
}

__global__ __launch_bounds__(256)
void zadj_kernel(const float* __restrict__ z, const float* __restrict__ adjW,
                 const float* __restrict__ adjb, float* __restrict__ zadj) {
  size_t i = (size_t)blockIdx.x * 256 + threadIdx.x;
  if (i >= (size_t)NB * EMB) return;
  int n = (int)(i / EMB), e = (int)(i % EMB);
  float s = adjb[e];
#pragma unroll
  for (int j = 0; j < LATD; ++j)
    s += z[(size_t)n * LATD + j] * adjW[(size_t)(EMB + j) * EMB + e];
  zadj[i] = s;
}

__global__ __launch_bounds__(256)
void addz_kernel(float* __restrict__ xq, const float* __restrict__ zadj,
                 bf16* __restrict__ xqb) {
  size_t i = (size_t)blockIdx.x * 256 + threadIdx.x;
  if (i >= (size_t)NB * EMB) return;
  float v = xq[i] + zadj[i];
  xq[i] = v;
  xqb[i] = (bf16)v;
}

__global__ __launch_bounds__(256)
void zero_tok_kernel(int* __restrict__ tok) {
  int i = blockIdx.x * 256 + threadIdx.x;
  if (i < NB) tok[i] = 0;
}

// One wave per (n, head, query). L=10 keys, HD=128 (4 dims/lane, wave32).
__global__ __launch_bounds__(256)
void attn_kernel(const float* __restrict__ qq, const float* __restrict__ kk,
                 const float* __restrict__ vv, bf16* __restrict__ o, int QL) {
  int wid = blockIdx.x * 8 + (threadIdx.x >> 5);
  int lane = threadIdx.x & 31;
  int q = wid % QL;
  int h = (wid / QL) % NHEAD;
  int n = wid / (QL * NHEAD);
  const float* qp = qq + ((size_t)(n * QL + q) * NHEAD + h) * HDIM + lane * 4;
  float q0 = qp[0], q1 = qp[1], q2 = qp[2], q3 = qp[3];
  float sc[LSEQ];
#pragma unroll
  for (int k = 0; k < LSEQ; ++k) {
    const float* kp = kk + ((size_t)(n * LSEQ + k) * NHEAD + h) * HDIM + lane * 4;
    float p = q0 * kp[0] + q1 * kp[1] + q2 * kp[2] + q3 * kp[3];
#pragma unroll
    for (int off = 16; off > 0; off >>= 1) p += __shfl_xor(p, off, 32);
    sc[k] = p * 0.08838834764831845f;  // 1/sqrt(128)
  }
  float m = sc[0];
#pragma unroll
  for (int k = 1; k < LSEQ; ++k) m = fmaxf(m, sc[k]);
  float den = 0.0f;
#pragma unroll
  for (int k = 0; k < LSEQ; ++k) { sc[k] = expf(sc[k] - m); den += sc[k]; }
  float inv = 1.0f / den;
  float o0 = 0, o1 = 0, o2 = 0, o3 = 0;
#pragma unroll
  for (int k = 0; k < LSEQ; ++k) {
    float w = sc[k] * inv;
    const float* vp = vv + ((size_t)(n * LSEQ + k) * NHEAD + h) * HDIM + lane * 4;
    o0 += w * vp[0]; o1 += w * vp[1]; o2 += w * vp[2]; o3 += w * vp[3];
  }
  bf16* op = o + (size_t)(n * QL + q) * EMB + h * HDIM + lane * 4;
  op[0] = (bf16)o0; op[1] = (bf16)o1; op[2] = (bf16)o2; op[3] = (bf16)o3;
}

// out = LN(a + b) * g + be ; one 256-thread block per row of 768.
__global__ __launch_bounds__(256)
void add_ln_kernel(const float* __restrict__ a, const float* __restrict__ b,
                   const float* __restrict__ g, const float* __restrict__ be,
                   float* __restrict__ outf, bf16* __restrict__ outb) {
  __shared__ float red[256];
  int row = blockIdx.x, t = threadIdx.x;
  size_t base = (size_t)row * EMB;
  float v[3];
  float s = 0.0f;
#pragma unroll
  for (int j = 0; j < 3; ++j) {
    int e = t + 256 * j;
    v[j] = a[base + e] + b[base + e];
    s += v[j];
  }
  red[t] = s; __syncthreads();
  for (int o = 128; o > 0; o >>= 1) { if (t < o) red[t] += red[t + o]; __syncthreads(); }
  float mean = red[0] * (1.0f / EMB);
  __syncthreads();
  float s2 = 0.0f;
#pragma unroll
  for (int j = 0; j < 3; ++j) { float d = v[j] - mean; s2 += d * d; }
  red[t] = s2; __syncthreads();
  for (int o = 128; o > 0; o >>= 1) { if (t < o) red[t] += red[t + o]; __syncthreads(); }
  float rstd = rsqrtf(red[0] * (1.0f / EMB) + LN_EPS);
#pragma unroll
  for (int j = 0; j < 3; ++j) {
    int e = t + 256 * j;
    float o = (v[j] - mean) * rstd * g[e] + be[e];
    if (outf) outf[base + e] = o;
    if (outb) outb[base + e] = (bf16)o;
  }
}

// pooled = mean_l enc_out ; mu/logvar = pooled @ W + b ; z = mu + eps*exp(lv/2)
__global__ __launch_bounds__(256)
void pool_latent_kernel(const float* __restrict__ eo, const float* __restrict__ muW,
                        const float* __restrict__ mub, const float* __restrict__ lvW,
                        const float* __restrict__ lvb, const float* __restrict__ eps,
                        float* __restrict__ mu_out, float* __restrict__ lv_out,
                        float* __restrict__ z) {
  __shared__ float pooled[EMB];
  __shared__ float smu[LATD], slv[LATD];
  int n = blockIdx.x, t = threadIdx.x;
#pragma unroll
  for (int j = 0; j < 3; ++j) {
    int e = t + 256 * j;
    float s = 0.0f;
    for (int l = 0; l < LSEQ; ++l) s += eo[((size_t)n * LSEQ + l) * EMB + e];
    pooled[e] = s * (1.0f / LSEQ);
  }
  __syncthreads();
  if (t < 2 * LATD) {
    int j = t % LATD;
    const float* W = (t < LATD) ? muW : lvW;
    float s = (t < LATD) ? mub[j] : lvb[j];
    for (int e = 0; e < EMB; ++e) s += pooled[e] * W[(size_t)e * LATD + j];
    if (t < LATD) { smu[j] = s; mu_out[(size_t)n * LATD + j] = s; }
    else          { slv[j] = s; lv_out[(size_t)n * LATD + j] = s; }
  }
  __syncthreads();
  if (t < LATD)
    z[(size_t)n * LATD + t] = smu[t] + eps[(size_t)n * LATD + t] * expf(0.5f * slv[t]);
}

// Row softmax over 8192 logits -> probs (strided into d_out), argmax -> tok.
__global__ __launch_bounds__(256)
void softmax_argmax_kernel(const float* __restrict__ logits,
                           float* __restrict__ probs, int* __restrict__ tok) {
  __shared__ float sval[256];
  __shared__ int sidx[256];
  int n = blockIdx.x, t = threadIdx.x;
  const float* lp = logits + (size_t)n * VOCAB;
  float m = -3.4e38f;
  int mi = 0;
#pragma unroll
  for (int j = 0; j < VOCAB / 256; ++j) {
    int e = t + 256 * j;
    float v = lp[e];
    if (v > m || (v == m && e < mi)) { m = v; mi = e; }
  }
  sval[t] = m; sidx[t] = mi; __syncthreads();
  for (int o = 128; o > 0; o >>= 1) {
    if (t < o) {
      float v2 = sval[t + o]; int i2 = sidx[t + o];
      if (v2 > sval[t] || (v2 == sval[t] && i2 < sidx[t])) { sval[t] = v2; sidx[t] = i2; }
    }
    __syncthreads();
  }
  float gm = sval[0];
  if (t == 0) tok[n] = sidx[0];
  __syncthreads();
  float ex[VOCAB / 256];
  float s = 0.0f;
#pragma unroll
  for (int j = 0; j < VOCAB / 256; ++j) { ex[j] = expf(lp[t + 256 * j] - gm); s += ex[j]; }
  sval[t] = s; __syncthreads();
  for (int o = 128; o > 0; o >>= 1) { if (t < o) sval[t] += sval[t + o]; __syncthreads(); }
  float inv = 1.0f / sval[0];
#pragma unroll
  for (int j = 0; j < VOCAB / 256; ++j)
    probs[(size_t)n * LSEQ * VOCAB + t + 256 * j] = ex[j] * inv;
}

// ---------------------------------------------------------------------------
// Host orchestration
// ---------------------------------------------------------------------------
extern "C" void kernel_launch(void* const* d_in, const int* in_sizes, int n_in,
                              void* d_out, int out_size, void* d_ws, size_t ws_size,
                              hipStream_t stream) {
  (void)in_sizes; (void)n_in; (void)out_size; (void)ws_size;
  // Inputs flattened in setup_inputs() dict insertion order (depth-first).
  const int*   x_in   = (const int*)d_in[0];
  const float* eps    = (const float*)d_in[1];
  const float* enc_we = (const float*)d_in[2];
  const float* enc_pe = (const float*)d_in[3];
  const float* Wq_e = (const float*)d_in[4];
  const float* Wk_e = (const float*)d_in[5];
  const float* Wv_e = (const float*)d_in[6];
  const float* Wo_e = (const float*)d_in[7];
  const float* bo_e = (const float*)d_in[8];
  const float* g1_e = (const float*)d_in[9];
  const float* b1_e = (const float*)d_in[10];
  const float* W1_e = (const float*)d_in[11];
  const float* bf1_e = (const float*)d_in[12];
  const float* W2_e = (const float*)d_in[13];
  const float* bf2_e = (const float*)d_in[14];
  const float* g2_e = (const float*)d_in[15];
  const float* b2_e = (const float*)d_in[16];
  const float* muW = (const float*)d_in[17];
  const float* mub = (const float*)d_in[18];
  const float* lvW = (const float*)d_in[19];
  const float* lvb = (const float*)d_in[20];
  const float* dec_we = (const float*)d_in[21];
  const float* dec_pe = (const float*)d_in[22];
  const float* adjW = (const float*)d_in[23];
  const float* adjb = (const float*)d_in[24];
  const float* Wq_d = (const float*)d_in[25];
  const float* Wk_d = (const float*)d_in[26];
  const float* Wv_d = (const float*)d_in[27];
  const float* Wo_d = (const float*)d_in[28];
  const float* bo_d = (const float*)d_in[29];
  const float* g1_d = (const float*)d_in[30];
  const float* b1_d = (const float*)d_in[31];
  const float* W1_d = (const float*)d_in[32];
  const float* bf1_d = (const float*)d_in[33];
  const float* W2_d = (const float*)d_in[34];
  const float* bf2_d = (const float*)d_in[35];
  const float* g2_d = (const float*)d_in[36];
  const float* b2_d = (const float*)d_in[37];
  const float* outW = (const float*)d_in[38];
  const float* outb = (const float*)d_in[39];

  float* recon = (float*)d_out;                               // (N,L,V)
  float* mu_out = recon + (size_t)NB * LSEQ * VOCAB;          // (N,LAT)
  float* lv_out = mu_out + (size_t)NB * LATD;                 // (N,LAT)

  // --- workspace arena (regions reused across phases) ---
  char* ws = (char*)d_ws;
  size_t off = 0;
  auto arena = [&](size_t bytes) -> void* {
    void* p = ws + off;
    off += (bytes + 255) & ~(size_t)255;
    return p;
  };
  float* h_f   = (float*)arena(NLE * 4);            // enc embed (later: decoder scratch)
  float* r1_f  = (float*)arena(NLE * 4);            // qq -> oproj -> ffn2out
  float* kk_f  = (float*)arena(NLE * 4);            // enc kk -> dec kk (persistent)
  float* vv_f  = (float*)arena(NLE * 4);            // enc vv -> dec vv (persistent)
  bf16*  r4_b  = (bf16*)arena(NLE * 2);             // h_b -> ao_b -> x1_b -> eo_b
  float* x1_f  = (float*)arena(NLE * 4);
  bf16*  ffn1b = (bf16*)arena((size_t)NLROWS * FFD * 2);  // (later: f1d + logits)
  float* eo_f  = (float*)arena(NLE * 4);
  // bf16 weight copies
  bf16* wq_e_b = (bf16*)arena((size_t)HDIM * HDIM * 2);
  bf16* wk_e_b = (bf16*)arena((size_t)HDIM * HDIM * 2);
  bf16* wv_e_b = (bf16*)arena((size_t)HDIM * HDIM * 2);
  bf16* wo_e_b = (bf16*)arena((size_t)EMB * EMB * 2);
  bf16* w1_e_b = (bf16*)arena((size_t)EMB * FFD * 2);
  bf16* w2_e_b = (bf16*)arena((size_t)FFD * EMB * 2);
  bf16* wq_d_b = (bf16*)arena((size_t)HDIM * HDIM * 2);
  bf16* wk_d_b = (bf16*)arena((size_t)HDIM * HDIM * 2);
  bf16* wv_d_b = (bf16*)arena((size_t)HDIM * HDIM * 2);
  bf16* wo_d_b = (bf16*)arena((size_t)EMB * EMB * 2);
  bf16* w1_d_b = (bf16*)arena((size_t)EMB * FFD * 2);
  bf16* w2_d_b = (bf16*)arena((size_t)FFD * EMB * 2);
  bf16* adj_b16 = (bf16*)arena((size_t)EMB * EMB * 2);   // top 768 rows of adj_W
  bf16* outW_b = (bf16*)arena((size_t)EMB * VOCAB * 2);
  float* z_ws  = (float*)arena((size_t)NB * LATD * 4);
  float* zadj  = (float*)arena((size_t)NB * EMB * 4);
  int*   tok   = (int*)arena((size_t)NB * 4);

  // decoder per-step buffers overlay the (then-dead) encoder h_f region
  char* dbase = (char*)h_f;
  size_t doff = 0;
  auto darena = [&](size_t bytes) -> void* {
    void* p = dbase + doff;
    doff += (bytes + 255) & ~(size_t)255;
    return p;
  };
  bf16*  xe_b  = (bf16*)darena((size_t)NB * EMB * 2);
  float* xq_f  = (float*)darena((size_t)NB * EMB * 4);
  bf16*  xq_b  = (bf16*)darena((size_t)NB * EMB * 2);
  float* qq_d  = (float*)darena((size_t)NB * EMB * 4);
  bf16*  o_b   = (bf16*)darena((size_t)NB * EMB * 2);
  float* opd_f = (float*)darena((size_t)NB * EMB * 4);
  float* xln_f = (float*)darena((size_t)NB * EMB * 4);
  bf16*  xln_b = (bf16*)darena((size_t)NB * EMB * 2);
  float* f2d_f = (float*)darena((size_t)NB * EMB * 4);
  bf16*  hq_b  = (bf16*)darena((size_t)NB * EMB * 2);
  // decoder ffn1 + logits overlay the encoder ffn1 region
  bf16*  f1d_b  = (bf16*)ffn1b;
  float* logits = (float*)((char*)ffn1b + (((size_t)NB * FFD * 2 + 255) & ~(size_t)255));

  auto f2b = [&](const float* s, bf16* d, size_t n) {
    hipLaunchKernelGGL(f2b_kernel, dim3((unsigned)((n + 255) / 256)), dim3(256), 0,
                       stream, s, d, n);
  };
  auto gemm_head = [&](const bf16* A, const bf16* B, float* C, int M) {
    hipLaunchKernelGGL((gemm_bf16_kernel<4, 2, false, float>), dim3(1, M / 128),
                       dim3(256), 0, stream, A, B, (const float*)nullptr, C, M, HDIM, HDIM);
  };
  auto gemm_f = [&](const bf16* A, const bf16* B, const float* bias, float* C,
                    int M, int K, int N) {
    hipLaunchKernelGGL((gemm_bf16_kernel<2, 4, false, float>), dim3(N / 256, M / 64),
                       dim3(256), 0, stream, A, B, bias, C, M, K, N);
  };
  auto gemm_relu_b = [&](const bf16* A, const bf16* B, const float* bias, bf16* C,
                         int M, int K, int N) {
    hipLaunchKernelGGL((gemm_bf16_kernel<2, 4, true, bf16>), dim3(N / 256, M / 64),
                       dim3(256), 0, stream, A, B, bias, C, M, K, N);
  };

  // --- weight conversions (once) ---
  f2b(Wq_e, wq_e_b, (size_t)HDIM * HDIM);
  f2b(Wk_e, wk_e_b, (size_t)HDIM * HDIM);
  f2b(Wv_e, wv_e_b, (size_t)HDIM * HDIM);
  f2b(Wo_e, wo_e_b, (size_t)EMB * EMB);
  f2b(W1_e, w1_e_b, (size_t)EMB * FFD);
  f2b(W2_e, w2_e_b, (size_t)FFD * EMB);
  f2b(Wq_d, wq_d_b, (size_t)HDIM * HDIM);
  f2b(Wk_d, wk_d_b, (size_t)HDIM * HDIM);
  f2b(Wv_d, wv_d_b, (size_t)HDIM * HDIM);
  f2b(Wo_d, wo_d_b, (size_t)EMB * EMB);
  f2b(W1_d, w1_d_b, (size_t)EMB * FFD);
  f2b(W2_d, w2_d_b, (size_t)FFD * EMB);
  f2b(adjW, adj_b16, (size_t)EMB * EMB);     // rows 0..767 are contiguous
  f2b(outW, outW_b, (size_t)EMB * VOCAB);

  // --- encoder ---
  hipLaunchKernelGGL(embed_enc_kernel, dim3((unsigned)(NLE / 256)), dim3(256), 0,
                     stream, x_in, enc_we, enc_pe, h_f, r4_b);
  gemm_head(r4_b, wq_e_b, r1_f, NLROWS * NHEAD);   // qq
  gemm_head(r4_b, wk_e_b, kk_f, NLROWS * NHEAD);   // kk
  gemm_head(r4_b, wv_e_b, vv_f, NLROWS * NHEAD);   // vv
  hipLaunchKernelGGL(attn_kernel, dim3(NB * NHEAD * LSEQ / 8), dim3(256), 0, stream,
                     r1_f, kk_f, vv_f, r4_b, LSEQ);              // -> ao (bf16)
  gemm_f(r4_b, wo_e_b, bo_e, r1_f, NLROWS, EMB, EMB);            // oproj
  hipLaunchKernelGGL(add_ln_kernel, dim3(NLROWS), dim3(256), 0, stream,
                     r1_f, h_f, g1_e, b1_e, x1_f, r4_b);         // x1 (f32 + bf16)
  gemm_relu_b(r4_b, w1_e_b, bf1_e, ffn1b, NLROWS, EMB, FFD);     // ffn1
  gemm_f(ffn1b, w2_e_b, bf2_e, r1_f, NLROWS, FFD, EMB);          // ffn2
  hipLaunchKernelGGL(add_ln_kernel, dim3(NLROWS), dim3(256), 0, stream,
                     r1_f, x1_f, g2_e, b2_e, eo_f, r4_b);        // enc_out (f32 + bf16)
  hipLaunchKernelGGL(pool_latent_kernel, dim3(NB), dim3(256), 0, stream,
                     eo_f, muW, mub, lvW, lvb, eps, mu_out, lv_out, z_ws);
  // decoder cross-attention K/V: computed once (enc_out is static)
  gemm_head(r4_b, wk_d_b, kk_f, NLROWS * NHEAD);
  gemm_head(r4_b, wv_d_b, vv_f, NLROWS * NHEAD);
  hipLaunchKernelGGL(zadj_kernel, dim3(NB * EMB / 256), dim3(256), 0, stream,
                     z_ws, adjW, adjb, zadj);                    // z@adjW[768:] + adj_b
  hipLaunchKernelGGL(zero_tok_kernel, dim3((NB + 255) / 256), dim3(256), 0, stream, tok);

  // --- autoregressive decoder: 10 greedy steps ---
  for (int p = 0; p < LSEQ; ++p) {
    hipLaunchKernelGGL(dec_embed_kernel, dim3(NB * EMB / 256), dim3(256), 0, stream,
                       tok, dec_we, dec_pe, p, xe_b);
    gemm_f(xe_b, adj_b16, (const float*)nullptr, xq_f, NB, EMB, EMB);
    hipLaunchKernelGGL(addz_kernel, dim3(NB * EMB / 256), dim3(256), 0, stream,
                       xq_f, zadj, xq_b);                        // xq (f32 + bf16)
    gemm_head(xq_b, wq_d_b, qq_d, NB * NHEAD);                   // qq
    hipLaunchKernelGGL(attn_kernel, dim3(NB * NHEAD / 8), dim3(256), 0, stream,
                       qq_d, kk_f, vv_f, o_b, 1);                // cross-attn
    gemm_f(o_b, wo_d_b, bo_d, opd_f, NB, EMB, EMB);              // oproj
    hipLaunchKernelGGL(add_ln_kernel, dim3(NB), dim3(256), 0, stream,
                       opd_f, xq_f, g1_d, b1_d, xln_f, xln_b);
    gemm_relu_b(xln_b, w1_d_b, bf1_d, f1d_b, NB, EMB, FFD);
    gemm_f(f1d_b, w2_d_b, bf2_d, f2d_f, NB, FFD, EMB);
    hipLaunchKernelGGL(add_ln_kernel, dim3(NB), dim3(256), 0, stream,
                       f2d_f, xln_f, g2_d, b2_d, (float*)nullptr, hq_b);
    gemm_f(hq_b, outW_b, outb, logits, NB, EMB, VOCAB);          // logits
    hipLaunchKernelGGL(softmax_argmax_kernel, dim3(NB), dim3(256), 0, stream,
                       logits, recon + (size_t)p * VOCAB, tok);  // probs + next tok
  }
}
